// MoETransformerEncoderLayer_66829691126405
// MI455X (gfx1250) — matrix operand order repaired and checked by reading, not compiled, not run on hardware
//
#include <hip/hip_runtime.h>
#include <hip/hip_bf16.h>

// ---------------------------------------------------------------------------
// Problem constants (from reference)
// ---------------------------------------------------------------------------
#define S_   1024
#define B_   2
#define E_   768
#define H_   12
#define NE_  8
#define HID_ 3072
#define T_   (S_ * B_)     // 2048 tokens
#define E3_  (3 * E_)      // 2304

typedef __attribute__((ext_vector_type(16))) _Float16 v16h;
typedef __attribute__((ext_vector_type(8)))  float    v8f;
typedef __attribute__((ext_vector_type(4)))  unsigned int u32x4;
typedef __attribute__((ext_vector_type(8)))  int      i32x8;
typedef __attribute__((ext_vector_type(4)))  int      i32x4;

union V16 { uint4 u[2]; v16h v; };
union U4H { uint4 u; _Float16 h[8]; };

// ---------------------------------------------------------------------------
// WMMA fragment loaders (wave32 layouts per CDNA5 ISA 7.12.2)
// ---------------------------------------------------------------------------
__device__ __forceinline__ v16h ld_a_frag(const _Float16* base, int ldk, int lane) {
    int m = lane & 15, half = lane >> 4;
    const _Float16* r = base + m * ldk;
    V16 x;
    x.u[0] = *(const uint4*)(r + half * 8);
    x.u[1] = *(const uint4*)(r + 16 + half * 8);
    return x.v;
}

__device__ __forceinline__ v16h ld_b_frag_t(const _Float16* baseT, int ldk, int lane) {
    int n = lane & 15, half = lane >> 4;
    const _Float16* r = baseT + n * ldk;
    V16 x;
    x.u[0] = *(const uint4*)(r + half * 8);
    x.u[1] = *(const uint4*)(r + 16 + half * 8);
    return x.v;
}

__device__ __forceinline__ float gelu_tanh(float x) {
    const float c = 0.7978845608028654f;
    float t = c * (x + 0.044715f * x * x * x);
    return 0.5f * x * (1.0f + tanhf(t));
}

// ---------------------------------------------------------------------------
// Tensor Data Mover: 2D f16 tile (tile_k contiguous elems x tile_n rows,
// row stride = stride_elems) from global to LDS at lds_byte_addr.
// Descriptor layout per CDNA5 ISA 8.3/8.4 (D# group0 + group1, groups2/3 zero).
// ---------------------------------------------------------------------------
__device__ __forceinline__ void tdm_load_2d_f16(const _Float16* gsrc,
                                                unsigned lds_byte_addr,
                                                unsigned tile_k, unsigned tile_n,
                                                unsigned stride_elems,
                                                unsigned tensor_d0, unsigned tensor_d1) {
    unsigned long long ga = (unsigned long long)(size_t)gsrc;
    u32x4 g0;
    g0[0] = 1u;                                   // count=1, user descriptor
    g0[1] = lds_byte_addr;                        // LDS byte address
    g0[2] = (unsigned)(ga & 0xffffffffu);         // global_addr[31:0]
    g0[3] = (unsigned)((ga >> 32) & 0x01ffffffu)  // global_addr[56:32]
          | (2u << 30);                           // type=2 ("image")
    i32x8 g1;
    g1[0] = (int)(1u << 16);                                      // data_size=1 (2B), wg_mask=0
    g1[1] = (int)((tensor_d0 & 0xffffu) << 16);                   // tensor_dim0[15:0]
    g1[2] = (int)(((tensor_d0 >> 16) & 0xffffu) |
                  ((tensor_d1 & 0xffffu) << 16));                 // dim0[31:16] | dim1[15:0]
    g1[3] = (int)(((tensor_d1 >> 16) & 0xffffu) |
                  ((tile_k & 0xffffu) << 16));                    // dim1[31:16] | tile_dim0
    g1[4] = (int)(tile_n & 0xffffu);                              // tile_dim1 (tile_dim2=0)
    g1[5] = (int)stride_elems;                                    // tensor_dim0_stride[31:0]
    g1[6] = 0;                                                    // stride[47:32]=0, dim1_stride lo=0
    g1[7] = 0;
    i32x4 z4 = {0, 0, 0, 0};
#if __clang_major__ >= 23
    i32x8 z8 = {0, 0, 0, 0, 0, 0, 0, 0};
    __builtin_amdgcn_tensor_load_to_lds(g0, g1, z4, z4, z8, 0);
#else
    __builtin_amdgcn_tensor_load_to_lds(g0, g1, z4, z4, 0);
#endif
}

// ---------------------------------------------------------------------------
// Generic WMMA GEMM with TDM-staged, double-buffered LDS tiles.
//   A: [M][K] f16 row-major.  Bn: [N][K] f16 (n-major; K contiguous).
//   MODE 1: Ch = f16(acc + bias)
//   MODE 2: Cf = resid + acc + bias
//   MODE 3: Ch = f16(gelu(acc + bias))
//   MODE 4: Cf += gates[m*NE+expert] * (acc + bias)
// Block: 256 threads (8 waves), tile 128x128, K-step 32.
// ---------------------------------------------------------------------------
#define BM 128
#define BN 128
#define BKK 32

template <int MODE>
__global__ __launch_bounds__(256)
void k_gemm(const _Float16* __restrict__ A, const _Float16* __restrict__ Bn,
            const float* __restrict__ bias,
            float* __restrict__ Cf, _Float16* __restrict__ Ch,
            const float* __restrict__ resid,
            const float* __restrict__ gates, int expert,
            int M, int N, int K) {
    __shared__ __align__(16) _Float16 sA[2][BM * BKK];   // [m][k] rows of 32
    __shared__ __align__(16) _Float16 sB[2][BN * BKK];   // [n][k] rows of 32

    const int tid  = threadIdx.x;
    const int lane = tid & 31;
    const int wave = tid >> 5;
    const int wrow = wave >> 1;   // 0..3
    const int wcol = wave & 1;    // 0..1
    const int m0 = blockIdx.y * BM;
    const int n0 = blockIdx.x * BN;

    v8f acc[2][4];
    const v8f vz = {0.f, 0.f, 0.f, 0.f, 0.f, 0.f, 0.f, 0.f};
    for (int i = 0; i < 2; ++i)
        for (int j = 0; j < 4; ++j) acc[i][j] = vz;

    const int ksteps = K / BKK;
    const bool issuer = (wave == 0);

    if (issuer) {   // prologue: stage tile 0 into buffer 0 via TDM
        tdm_load_2d_f16(A  + (size_t)m0 * K, (unsigned)(size_t)&sA[0][0],
                        BKK, BM, (unsigned)K, (unsigned)K, (unsigned)M);
        tdm_load_2d_f16(Bn + (size_t)n0 * K, (unsigned)(size_t)&sB[0][0],
                        BKK, BN, (unsigned)K, (unsigned)K, (unsigned)N);
    }

    for (int kt = 0; kt < ksteps; ++kt) {
        const int cur = kt & 1;
        if (issuer) {
            if (kt + 1 < ksteps) {  // stage next tile into the other buffer
                tdm_load_2d_f16(A  + (size_t)m0 * K + (kt + 1) * BKK,
                                (unsigned)(size_t)&sA[cur ^ 1][0],
                                BKK, BM, (unsigned)K, (unsigned)K, (unsigned)M);
                tdm_load_2d_f16(Bn + (size_t)n0 * K + (kt + 1) * BKK,
                                (unsigned)(size_t)&sB[cur ^ 1][0],
                                BKK, BN, (unsigned)K, (unsigned)K, (unsigned)N);
                __builtin_amdgcn_s_wait_tensorcnt(2);  // tile kt resident
            } else {
                __builtin_amdgcn_s_wait_tensorcnt(0);
            }
        }
        __syncthreads();

        v16h af[2], bf[4];
        #pragma unroll
        for (int i = 0; i < 2; ++i)
            af[i] = ld_a_frag(&sA[cur][(wrow * 32 + i * 16) * BKK], BKK, lane);
        #pragma unroll
        for (int j = 0; j < 4; ++j)
            bf[j] = ld_b_frag_t(&sB[cur][(wcol * 64 + j * 16) * BKK], BKK, lane);

        #pragma unroll
        for (int i = 0; i < 2; ++i)
            #pragma unroll
            for (int j = 0; j < 4; ++j)
                acc[i][j] = __builtin_amdgcn_wmma_f32_16x16x32_f16(
                    false, af[i], false, bf[j], (short)0, acc[i][j], false, false);
        __syncthreads();
    }

    // Epilogue (straight-line per MODE). C/D layout: lane n = lane&15,
    // rows m = (lane>>4)*8 + r.
    const int half = lane >> 4, lcol = lane & 15;
    #pragma unroll
    for (int i = 0; i < 2; ++i) {
        #pragma unroll
        for (int j = 0; j < 4; ++j) {
            int mb = m0 + wrow * 32 + i * 16 + half * 8;
            int nb = n0 + wcol * 64 + j * 16 + lcol;
            float bv = bias[nb];
            #pragma unroll
            for (int r = 0; r < 8; ++r) {
                int m = mb + r;
                float v = acc[i][j][r] + bv;
                size_t idx = (size_t)m * N + nb;
                if (MODE == 1) {
                    Ch[idx] = (_Float16)v;
                } else if (MODE == 2) {
                    Cf[idx] = resid[idx] + v;
                } else if (MODE == 3) {
                    Ch[idx] = (_Float16)gelu_tanh(v);
                } else { // MODE == 4
                    Cf[idx] += gates[(size_t)m * NE_ + expert] * v;
                }
            }
        }
    }
}

// ---------------------------------------------------------------------------
// Flash-style attention: one wave per (b, h, 16-query tile).
// qkv16: [S,B,3E] f16 rows (q|k|v). o16: [S,B,E] f16.
// ---------------------------------------------------------------------------
__global__ __launch_bounds__(32)
void k_attn(const _Float16* __restrict__ qkv, _Float16* __restrict__ o16) {
    __shared__ __align__(16) _Float16 sK[32 * 64];   // [key][d]
    __shared__ __align__(16) _Float16 sVt[64 * 32];  // [d][key]
    __shared__ __align__(16) _Float16 sP[16 * 32];   // [m][key_local]

    const int lane = threadIdx.x;
    const int s0 = blockIdx.x * 16;
    const int h  = blockIdx.y;
    const int b  = blockIdx.z;
    const int half = lane >> 4, lcol = lane & 15;

    v16h qf[2];
    {
        int m = lcol;
        const _Float16* qr = qkv + ((size_t)(s0 + m) * B_ + b) * E3_ + h * 64;
        #pragma unroll
        for (int f = 0; f < 2; ++f) {
            V16 x;
            x.u[0] = *(const uint4*)(qr + f * 32 + half * 8);
            x.u[1] = *(const uint4*)(qr + f * 32 + 16 + half * 8);
            qf[f] = x.v;
        }
    }

    const v8f vz = {0.f, 0.f, 0.f, 0.f, 0.f, 0.f, 0.f, 0.f};
    v8f accO[4] = {vz, vz, vz, vz};
    float mrow[8], lrow[8];
    #pragma unroll
    for (int i = 0; i < 8; ++i) { mrow[i] = -1e30f; lrow[i] = 0.f; }

    for (int kb = 0; kb < S_ / 32; ++kb) {
        {
            int r = lane;
            const _Float16* kp = qkv + ((size_t)(kb * 32 + r) * B_ + b) * E3_ + E_ + h * 64;
            const _Float16* vp = kp + E_;
            #pragma unroll
            for (int j = 0; j < 8; ++j) {
                uint4 kv = *(const uint4*)(kp + j * 8);
                *(uint4*)(sK + r * 64 + j * 8) = kv;
                U4H vv; vv.u = *(const uint4*)(vp + j * 8);
                #pragma unroll
                for (int e = 0; e < 8; ++e)
                    sVt[(j * 8 + e) * 32 + r] = vv.h[e];
            }
        }
        __syncthreads();

        v8f sc[2];
        #pragma unroll
        for (int t = 0; t < 2; ++t) {
            v8f s = vz;
            #pragma unroll
            for (int f = 0; f < 2; ++f) {
                v16h bk = ld_b_frag_t(sK + t * 16 * 64 + f * 32, 64, lane);
                s = __builtin_amdgcn_wmma_f32_16x16x32_f16(
                        false, qf[f], false, bk, (short)0, s, false, false);
            }
            sc[t] = s * 0.125f;   // HD^-0.5, HD=64
        }

        float pv0[8], pv1[8];
        #pragma unroll
        for (int i = 0; i < 8; ++i) {
            float tm = fmaxf(sc[0][i], sc[1][i]);
            for (int off = 8; off; off >>= 1)
                tm = fmaxf(tm, __shfl_xor(tm, off, 32));
            float mn = fmaxf(mrow[i], tm);
            float alpha = __expf(mrow[i] - mn);
            float p0 = __expf(sc[0][i] - mn);
            float p1 = __expf(sc[1][i] - mn);
            float rs = p0 + p1;
            for (int off = 8; off; off >>= 1)
                rs += __shfl_xor(rs, off, 32);
            lrow[i] = lrow[i] * alpha + rs;
            mrow[i] = mn;
            #pragma unroll
            for (int dt = 0; dt < 4; ++dt) accO[dt][i] *= alpha;
            pv0[i] = p0; pv1[i] = p1;
        }

        #pragma unroll
        for (int i = 0; i < 8; ++i) {
            sP[(half * 8 + i) * 32 + lcol]      = (_Float16)pv0[i];
            sP[(half * 8 + i) * 32 + 16 + lcol] = (_Float16)pv1[i];
        }
        __syncthreads();

        v16h pf = ld_a_frag(sP, 32, lane);
        #pragma unroll
        for (int dt = 0; dt < 4; ++dt) {
            v16h vf = ld_b_frag_t(sVt + dt * 16 * 32, 32, lane);
            accO[dt] = __builtin_amdgcn_wmma_f32_16x16x32_f16(
                           false, pf, false, vf, (short)0, accO[dt], false, false);
        }
        __syncthreads();
    }

    for (int dt = 0; dt < 4; ++dt) {
        #pragma unroll
        for (int i = 0; i < 8; ++i) {
            int m = half * 8 + i;
            float v = accO[dt][i] / lrow[i];
            o16[((size_t)(s0 + m) * B_ + b) * E_ + h * 64 + dt * 16 + lcol] = (_Float16)v;
        }
    }
}

// ---------------------------------------------------------------------------
// LayerNorm: one block (256 thr) per token row of length E=768.
// ---------------------------------------------------------------------------
__global__ __launch_bounds__(256)
void k_ln(const float* __restrict__ x, const float* __restrict__ g,
          const float* __restrict__ bb, float* __restrict__ y32,
          _Float16* __restrict__ y16) {
    __shared__ float s1[8], s2[8];
    const size_t t = blockIdx.x;
    const float* row = x + t * E_;
    float a0 = 0.f, a1 = 0.f;
    for (int e = threadIdx.x; e < E_; e += 256) {
        float v = row[e]; a0 += v; a1 += v * v;
    }
    for (int off = 16; off; off >>= 1) {
        a0 += __shfl_xor(a0, off, 32);
        a1 += __shfl_xor(a1, off, 32);
    }
    if ((threadIdx.x & 31) == 0) {
        s1[threadIdx.x >> 5] = a0; s2[threadIdx.x >> 5] = a1;
    }
    __syncthreads();
    float mu = 0.f, qq = 0.f;
    #pragma unroll
    for (int i = 0; i < 8; ++i) { mu += s1[i]; qq += s2[i]; }
    mu *= (1.f / E_);
    qq = qq * (1.f / E_) - mu * mu;
    float inv = rsqrtf(qq + 1e-5f);
    for (int e = threadIdx.x; e < E_; e += 256) {
        float v = (row[e] - mu) * inv * g[e] + bb[e];
        if (y32) y32[t * E_ + e] = v;
        y16[t * E_ + e] = (_Float16)v;
    }
}

// ---------------------------------------------------------------------------
// Gating: one wave per token; logits, softmax(8), top-2, renorm.
// ---------------------------------------------------------------------------
__global__ __launch_bounds__(256)
void k_gate(const float* __restrict__ x, const float* __restrict__ gw,
            const float* __restrict__ gb, float* __restrict__ gates) {
    const int t = blockIdx.x * 8 + (threadIdx.x >> 5);
    const int lane = threadIdx.x & 31;
    float lg[NE_];
    for (int n = 0; n < NE_; ++n) {
        float p = 0.f;
        for (int e = lane; e < E_; e += 32)
            p += x[(size_t)t * E_ + e] * gw[n * E_ + e];
        for (int off = 16; off; off >>= 1) p += __shfl_xor(p, off, 32);
        lg[n] = p + gb[n];
    }
    float mx = lg[0];
    #pragma unroll
    for (int n = 1; n < NE_; ++n) mx = fmaxf(mx, lg[n]);
    float pr[NE_], sum = 0.f;
    #pragma unroll
    for (int n = 0; n < NE_; ++n) { pr[n] = __expf(lg[n] - mx); sum += pr[n]; }
    #pragma unroll
    for (int n = 0; n < NE_; ++n) pr[n] /= sum;
    int i1 = 0;
    #pragma unroll
    for (int n = 1; n < NE_; ++n) if (pr[n] > pr[i1]) i1 = n;
    int i2 = (i1 == 0) ? 1 : 0;
    #pragma unroll
    for (int n = 0; n < NE_; ++n) if (n != i1 && pr[n] > pr[i2]) i2 = n;
    float s2v = pr[i1] + pr[i2];
    if (lane < NE_) {
        float gv = (lane == i1) ? pr[i1] / s2v : (lane == i2) ? pr[i2] / s2v : 0.f;
        gates[(size_t)t * NE_ + lane] = gv;
    }
}

// ---------------------------------------------------------------------------
// Small utility kernels
// ---------------------------------------------------------------------------
__global__ __launch_bounds__(256)
void k_cvt(const float* __restrict__ x, _Float16* __restrict__ y, int n) {
    int i = blockIdx.x * 256 + threadIdx.x;
    if (i < n) y[i] = (_Float16)x[i];
}

// Per-expert transpose+convert: in [R][C] f32 -> out [C][R] f16; expert = blockIdx.y
__global__ __launch_bounds__(256)
void k_transpose(const float* __restrict__ in, _Float16* __restrict__ out, int R, int C) {
    const float* srcp = in + (size_t)blockIdx.y * R * C;
    _Float16* dstp = out + (size_t)blockIdx.y * R * C;
    int i = blockIdx.x * 256 + threadIdx.x;
    if (i < R * C) {
        int r = i / C, c = i % C;
        dstp[(size_t)c * R + r] = (_Float16)srcp[i];
    }
}

__global__ __launch_bounds__(256)
void k_add(const float* __restrict__ a, const float* __restrict__ bsrc,
           float* __restrict__ o, int n) {
    int i = blockIdx.x * 256 + threadIdx.x;
    if (i < n) o[i] = a[i] + bsrc[i];
}

// ---------------------------------------------------------------------------
// Host launcher
// ---------------------------------------------------------------------------
extern "C" void kernel_launch(void* const* d_in, const int* in_sizes, int n_in,
                              void* d_out, int out_size, void* d_ws, size_t ws_size,
                              hipStream_t stream) {
    const float* src        = (const float*)d_in[0];
    const float* ln1_g      = (const float*)d_in[1];
    const float* ln1_b      = (const float*)d_in[2];
    const float* ln2_g      = (const float*)d_in[3];
    const float* ln2_b      = (const float*)d_in[4];
    const float* in_proj_w  = (const float*)d_in[5];
    const float* in_proj_b  = (const float*)d_in[6];
    const float* out_proj_w = (const float*)d_in[7];
    const float* out_proj_b = (const float*)d_in[8];
    const float* gate_w     = (const float*)d_in[9];
    const float* gate_b     = (const float*)d_in[10];
    const float* w1         = (const float*)d_in[11];
    const float* b1         = (const float*)d_in[12];
    const float* w2         = (const float*)d_in[13];
    const float* b2         = (const float*)d_in[14];

    float* out_x  = (float*)d_out;                     // [S,B,E]
    float* gates  = (float*)d_out + (size_t)T_ * E_;   // [T,NE]

    // Workspace carve-up (256B aligned)
    char* wsp = (char*)d_ws;
    size_t off = 0;
    auto alloc = [&](size_t bytes) {
        void* p = wsp + off;
        off = (off + bytes + 255) & ~(size_t)255;
        return p;
    };
    // All B matrices stored n-major [N][K] (K contiguous) for TDM staging.
    _Float16* WINP  = (_Float16*)alloc((size_t)E3_ * E_ * 2);        // [3E][E]
    _Float16* WOUT  = (_Float16*)alloc((size_t)E_ * E_ * 2);         // [E][E]
    _Float16* W1H   = (_Float16*)alloc((size_t)NE_ * HID_ * E_ * 2); // [NE][HID][E]
    _Float16* W2H   = (_Float16*)alloc((size_t)NE_ * E_ * HID_ * 2); // [NE][E][HID]
    _Float16* X2A   = (_Float16*)alloc((size_t)T_ * E_ * 2);
    _Float16* QKV16 = (_Float16*)alloc((size_t)T_ * E3_ * 2);
    _Float16* O16   = (_Float16*)alloc((size_t)T_ * E_ * 2);
    float*    SRC2  = (float*)   alloc((size_t)T_ * E_ * 4);
    _Float16* X2B16 = (_Float16*)alloc((size_t)T_ * E_ * 2);
    float*    X2B32 = (float*)   alloc((size_t)T_ * E_ * 4);
    _Float16* H16   = (_Float16*)alloc((size_t)T_ * HID_ * 2);
    float*    MOE   = (float*)   alloc((size_t)T_ * E_ * 4);
    (void)ws_size; (void)n_in; (void)in_sizes; (void)out_size;

    // 1) Weight prep: in_proj_w/out_proj_w are already [N][K]; w1/w2 transposed.
    k_cvt<<<dim3((E3_ * E_ + 255) / 256), 256, 0, stream>>>(in_proj_w, WINP, E3_ * E_);
    k_cvt<<<dim3((E_ * E_ + 255) / 256), 256, 0, stream>>>(out_proj_w, WOUT, E_ * E_);
    k_transpose<<<dim3((E_ * HID_ + 255) / 256, NE_), 256, 0, stream>>>(w1, W1H, E_, HID_);
    k_transpose<<<dim3((E_ * HID_ + 255) / 256, NE_), 256, 0, stream>>>(w2, W2H, HID_, E_);

    // 2) LN1 -> f16 activations
    k_ln<<<dim3(T_), 256, 0, stream>>>(src, ln1_g, ln1_b, nullptr, X2A);

    // 3) QKV GEMM: [T,E] x [E,3E] -> f16 [T,3E]
    k_gemm<1><<<dim3(E3_ / BN, T_ / BM), 256, 0, stream>>>(
        X2A, WINP, in_proj_b, nullptr, QKV16, nullptr, nullptr, 0,
        T_, E3_, E_);

    // 4) Attention
    k_attn<<<dim3(S_ / 16, H_, B_), 32, 0, stream>>>(QKV16, O16);

    // 5) out_proj + residual: SRC2 = src + O*Wo + bo
    k_gemm<2><<<dim3(E_ / BN, T_ / BM), 256, 0, stream>>>(
        O16, WOUT, out_proj_b, SRC2, nullptr, src, nullptr, 0,
        T_, E_, E_);

    // 6) LN2
    k_ln<<<dim3(T_), 256, 0, stream>>>(SRC2, ln2_g, ln2_b, X2B32, X2B16);

    // 7) Gating (writes gates portion of d_out)
    k_gate<<<dim3(T_ / 8), 256, 0, stream>>>(X2B32, gate_w, gate_b, gates);

    // 8) MoE accumulation buffer
    hipMemsetAsync(MOE, 0, (size_t)T_ * E_ * 4, stream);

    // 9) Dense all-expert MoE, gate-combined in the w2 epilogue
    for (int n = 0; n < NE_; ++n) {
        k_gemm<3><<<dim3(HID_ / BN, T_ / BM), 256, 0, stream>>>(
            X2B16, W1H + (size_t)n * E_ * HID_, b1 + (size_t)n * HID_,
            nullptr, H16, nullptr, nullptr, 0, T_, HID_, E_);
        k_gemm<4><<<dim3(E_ / BN, T_ / BM), 256, 0, stream>>>(
            H16, W2H + (size_t)n * HID_ * E_, b2 + (size_t)n * E_,
            MOE, nullptr, nullptr, gates, n, T_, E_, HID_);
    }

    // 10) Final residual add
    k_add<<<dim3((T_ * E_ + 255) / 256), 256, 0, stream>>>(MOE, src, out_x, T_ * E_);
}
